// Net_57836029608014
// MI455X (gfx1250) — matrix operand order repaired and checked by reading, not compiled, not run on hardware
//
#include <hip/hip_runtime.h>
#include <hip/hip_bf16.h>

#define N_NODES 32768
#define E_EDGES 65536
#define B_GRAPHS 1024
#define D_DIM 64
#define R_RANK 512
#define FEAT_IN 13
#define EF_DIM 5
#define NT_TYPES 5
#define NA_TYPES 7
#define F2_N 8192
#define F3_N 6144
#define F4_N 4096

typedef __attribute__((ext_vector_type(16))) _Float16 v16h;
typedef __attribute__((ext_vector_type(8)))  float    v8f;
typedef _Float16 half_t;

// ---------------------------------------------------------------------------
// helpers
// ---------------------------------------------------------------------------
__device__ __forceinline__ float sigmoidf_(float x) { return 1.0f / (1.0f + expf(-x)); }

__device__ __forceinline__ v8f wmma_f16(v16h a, v16h b, v8f c) {
  // v_wmma_f32_16x16x32_f16: D = A(16x32 f16) * B(32x16 f16) + C(16x16 f32)
  return __builtin_amdgcn_wmma_f32_16x16x32_f16(false, a, false, b, (short)0, c, false, false);
}

// A fragment (16x32 f16) from a row-major [16 x ldk] f16 tile (LDS or generic).
// ISA layout: lane<16: M=lane, K = {0..7, 16..23}; lane>=16: M=lane-16, K += 8.
__device__ __forceinline__ v16h a_frag(const half_t* A, int ldk, int kbase, int lane) {
  int m  = lane & 15;
  int kh = (lane < 16) ? 0 : 8;
  _Float16 t[16] __attribute__((aligned(32)));
#pragma unroll
  for (int i = 0; i < 8; ++i) {
    t[i]     = A[m * ldk + kbase + kh + i];
    t[8 + i] = A[m * ldk + kbase + 16 + kh + i];
  }
  return *(const v16h*)t;
}

// float max-key for atomic segment-max (order-preserving uint encoding)
__device__ __forceinline__ unsigned fkey(float f) {
  unsigned u = __float_as_uint(f);
  return (u & 0x80000000u) ? ~u : (u | 0x80000000u);
}
__device__ __forceinline__ float fkey_inv(unsigned k) {
  return (k & 0x80000000u) ? __uint_as_float(k & 0x7FFFFFFFu) : __uint_as_float(~k);
}

// ---------------------------------------------------------------------------
// generic small kernels
// ---------------------------------------------------------------------------
__global__ void fill_kernel(float* p, float v, int n) {
  int i = blockIdx.x * blockDim.x + threadIdx.x;
  if (i < n) p[i] = v;
}

__global__ void count_kernel(const int* __restrict__ ei, float* __restrict__ cnt) {
  int e = blockIdx.x * blockDim.x + threadIdx.x;
  if (e < E_EDGES) atomicAdd(&cnt[ei[E_EDGES + e]], 1.0f);
}

// atoms = argmax(x[:, :7]); out = h = relu(x @ lin0_W + lin0_b); outH = f16(out)
__global__ __launch_bounds__(256) void prep_kernel(const float* __restrict__ x,
                                                   const float* __restrict__ W,
                                                   const float* __restrict__ b,
                                                   float* __restrict__ out, float* __restrict__ h,
                                                   half_t* __restrict__ outH, int* __restrict__ atoms) {
  __shared__ float xr[4][FEAT_IN];
  int tid = threadIdx.x, nl = tid >> 6, d = tid & 63;
  int n = blockIdx.x * 4 + nl;
  if (d < FEAT_IN) xr[nl][d] = x[n * FEAT_IN + d];
  __syncthreads();
  if (d == 0) {
    int best = 0; float bv = xr[nl][0];
    for (int k = 1; k < NA_TYPES; ++k) if (xr[nl][k] > bv) { bv = xr[nl][k]; best = k; }
    atoms[n] = best;
  }
  float acc = b[d];
  for (int k = 0; k < FEAT_IN; ++k) acc += xr[nl][k] * W[k * 64 + d];
  acc = fmaxf(acc, 0.0f);
  out[n * 64 + d] = acc; h[n * 64 + d] = acc; outH[n * 64 + d] = (half_t)acc;
}

// h1 = relu(edge_attr @ nn1_W + nn1_b) as f16  (fixed across conv iterations)
__global__ __launch_bounds__(256) void edge_h1_kernel(const float* __restrict__ ea,
                                                      const float* __restrict__ W,
                                                      const float* __restrict__ b,
                                                      half_t* __restrict__ h1H) {
  __shared__ float er[2][EF_DIM];
  int tid = threadIdx.x, el = tid >> 7, c = tid & 127;
  int e = blockIdx.x * 2 + el;
  if (c < EF_DIM) er[el][c] = ea[e * EF_DIM + c];
  __syncthreads();
  float acc = b[c];
  for (int k = 0; k < EF_DIM; ++k) acc += er[el][k] * W[k * 128 + c];
  h1H[e * 128 + c] = (half_t)fmaxf(acc, 0.0f);
}

// Pack T' (K=8256 x N=64, f16) in WMMA-B lane layout:
// dst[((kc*4+fb)*32+lane)*16+i] = T'[kc*32 + (i&7) + (i>>3)*16 + (lane<16?0:8), fb*16 + lane%16]
// T'[c*64+d, f] = nn2_W[c, d*64+f];  K rows 8192..8255: nn2_b[d*64+f]
__global__ void pack_T_kernel(const float* __restrict__ nn2W, const float* __restrict__ nn2b,
                              half_t* __restrict__ dst) {
  int idx = blockIdx.x * blockDim.x + threadIdx.x;
  if (idx >= 258 * 4 * 32 * 16) return;
  int i = idx & 15, lane = (idx >> 4) & 31, fb = (idx >> 9) & 3, kc = idx >> 11;
  int n = fb * 16 + (lane & 15);
  int k = kc * 32 + (i & 7) + ((i >> 3) << 4) + ((lane < 16) ? 0 : 8);
  float v;
  if (k < 8192) { int c = k >> 6, d = k & 63; v = nn2W[c * 4096 + d * 64 + n]; }
  else          { int d = k - 8192;           v = nn2b[d * 64 + n]; }
  dst[idx] = (half_t)v;
}

// Generic pack of row-major [K,N] f32 into WMMA-B lane layout f16
__global__ void pack_B_kernel(const float* __restrict__ src, int K, int Ncols,
                              half_t* __restrict__ dst) {
  int total = (K / 32) * (Ncols / 16) * 512;
  int idx = blockIdx.x * blockDim.x + threadIdx.x;
  if (idx >= total) return;
  int i = idx & 15, lane = (idx >> 4) & 31, tile = idx >> 9;
  int nbc = Ncols / 16;
  int nb = tile % nbc, kc = tile / nbc;
  int k = kc * 32 + (i & 7) + ((i >> 3) << 4) + ((lane < 16) ? 0 : 8);
  int n = nb * 16 + (lane & 15);
  dst[idx] = (half_t)src[k * Ncols + n];
}

// ---------------------------------------------------------------------------
// NNConv fused edge-message GEMM:  msg[16e, 64f] = A'(16 x 8256) @ T'(8256 x 64)
// A'[e, c*64+d] = h1[e,c] * x_src[e,d]  (rank-1 outer product, built in regs)
// Branch-free K loop, ping-pong double-buffered B fragments, loop-invariant
// A base patterns held in registers.
// ---------------------------------------------------------------------------
__device__ __forceinline__ void loadB4(v16h* b, const half_t* __restrict__ Tpack, int kc, int lane) {
  const half_t* bp = Tpack + (size_t)(kc * 4 * 32 + lane) * 16;
#pragma unroll
  for (int fb = 0; fb < 4; ++fb) b[fb] = *(const v16h*)(bp + fb * 32 * 16);
}

__global__ __launch_bounds__(128) void conv_msg_kernel(const half_t* __restrict__ outH,
                                                       const half_t* __restrict__ h1H,
                                                       const half_t* __restrict__ Tpack,
                                                       const int* __restrict__ edge_index,
                                                       float* __restrict__ s) {
  __shared__ half_t xs[4][16][64];
  __shared__ half_t h1s[4][16][128];
  __shared__ int tgts[4][16];
  const int tid = threadIdx.x, w = tid >> 5, lane = tid & 31;
  const int ebase = (blockIdx.x * 4 + w) * 16;
  {
    int r = lane >> 1, part = lane & 1;
    int e = ebase + r;
    int src = edge_index[e];
    if (part == 0) tgts[w][r] = edge_index[E_EDGES + e];
    const half_t* xsrc = outH + (size_t)src * 64 + part * 32;
    for (int j = 0; j < 32; ++j) xs[w][r][part * 32 + j] = xsrc[j];
    const half_t* hsrc = h1H + (size_t)e * 128 + part * 64;
    for (int j = 0; j < 64; ++j) h1s[w][r][part * 64 + j] = hsrc[j];
  }
  __syncthreads();

  const int m = lane & 15;
  const int kh = (lane < 16) ? 0 : 8;

  // Loop-invariant A base patterns (lane touches only these 32 halfs of xs row)
  _Float16 t0[16] __attribute__((aligned(32)));
  _Float16 t1[16] __attribute__((aligned(32)));
#pragma unroll
  for (int i = 0; i < 8; ++i) {
    t0[i]     = xs[w][m][kh + i];
    t0[8 + i] = xs[w][m][16 + kh + i];
    t1[i]     = xs[w][m][32 + kh + i];
    t1[8 + i] = xs[w][m][48 + kh + i];
  }
  const v16h ax0 = *(const v16h*)t0;
  const v16h ax1 = *(const v16h*)t1;
  const half_t* h1row = &h1s[w][m][0];

  v8f acc0 = {}, acc1 = {}, acc2 = {}, acc3 = {};
  v16h bb0[4], bb1[4];
  loadB4(bb0, Tpack, 0, lane);

  // Main loop: c = 0..127, chunks (2c, 2c+1); h1 scalars fetched 4 at a time.
  for (int cg = 0; cg < 128; cg += 4) {
    union { unsigned u32[2]; _Float16 hx[4]; } h4;
    *(uint2*)h4.u32 = *(const uint2*)(h1row + cg);
#pragma unroll
    for (int cc = 0; cc < 4; ++cc) {
      const _Float16 h1v = h4.hx[cc];
      const int kc0 = (cg + cc) * 2;
      // chunk kc0 (uses bb0), prefetch kc0+1 into bb1
      loadB4(bb1, Tpack, kc0 + 1, lane);
      {
        v16h a = ax0 * h1v;
        acc0 = wmma_f16(a, bb0[0], acc0);
        acc1 = wmma_f16(a, bb0[1], acc1);
        acc2 = wmma_f16(a, bb0[2], acc2);
        acc3 = wmma_f16(a, bb0[3], acc3);
      }
      // chunk kc0+1 (uses bb1), prefetch kc0+2 into bb0
      loadB4(bb0, Tpack, kc0 + 2, lane);
      {
        v16h a = ax1 * h1v;
        acc0 = wmma_f16(a, bb1[0], acc0);
        acc1 = wmma_f16(a, bb1[1], acc1);
        acc2 = wmma_f16(a, bb1[2], acc2);
        acc3 = wmma_f16(a, bb1[3], acc3);
      }
    }
  }
  // Bias epilogue: chunks 256 (in bb0) and 257; A' = x_src against packed nn2_b.
  loadB4(bb1, Tpack, 257, lane);
  acc0 = wmma_f16(ax0, bb0[0], acc0);
  acc1 = wmma_f16(ax0, bb0[1], acc1);
  acc2 = wmma_f16(ax0, bb0[2], acc2);
  acc3 = wmma_f16(ax0, bb0[3], acc3);
  acc0 = wmma_f16(ax1, bb1[0], acc0);
  acc1 = wmma_f16(ax1, bb1[1], acc1);
  acc2 = wmma_f16(ax1, bb1[2], acc2);
  acc3 = wmma_f16(ax1, bb1[3], acc3);

  // C layout: VGPR r -> M = r + (lane<16?0:8), N = lane%16 (+16*fb)
  const int nlo = lane & 15;
  const int mofs = (lane < 16) ? 0 : 8;
#pragma unroll
  for (int r = 0; r < 8; ++r) {
    int row = r + mofs;
    float* srow = s + (size_t)tgts[w][row] * 64;
    atomicAdd(srow + 0 * 16 + nlo, acc0[r]);
    atomicAdd(srow + 1 * 16 + nlo, acc1[r]);
    atomicAdd(srow + 2 * 16 + nlo, acc2[r]);
    atomicAdd(srow + 3 * 16 + nlo, acc3[r]);
  }
}

// m = relu(out@root_W + root_b + s/max(cnt,1)); h = GRU(m,h); out = h; outH = f16(h)
__global__ __launch_bounds__(256) void gru_kernel(const float* __restrict__ rootW,
                                                  const float* __restrict__ rootb,
                                                  const float* __restrict__ Wih,
                                                  const float* __restrict__ Whh,
                                                  const float* __restrict__ bih,
                                                  const float* __restrict__ bhh,
                                                  const float* __restrict__ s,
                                                  const float* __restrict__ cnt,
                                                  float* __restrict__ out, float* __restrict__ h,
                                                  half_t* __restrict__ outH) {
  __shared__ float orow[4][64], hrow[4][64], mrow[4][64];
  int tid = threadIdx.x, nl = tid >> 6, d = tid & 63;
  int n = blockIdx.x * 4 + nl;
  orow[nl][d] = out[n * 64 + d];
  hrow[nl][d] = h[n * 64 + d];
  __syncthreads();
  float acc = rootb[d] + s[n * 64 + d] / fmaxf(cnt[n], 1.0f);
  for (int k = 0; k < 64; ++k) acc += orow[nl][k] * rootW[k * 64 + d];
  float m = fmaxf(acc, 0.0f);
  mrow[nl][d] = m;
  __syncthreads();
  float gir = bih[d], giz = bih[64 + d], gin = bih[128 + d];
  float ghr = bhh[d], ghz = bhh[64 + d], ghn = bhh[128 + d];
  for (int k = 0; k < 64; ++k) {
    float mk = mrow[nl][k], hk = hrow[nl][k];
    gir += mk * Wih[k * 192 + d];       ghr += hk * Whh[k * 192 + d];
    giz += mk * Wih[k * 192 + 64 + d];  ghz += hk * Whh[k * 192 + 64 + d];
    gin += mk * Wih[k * 192 + 128 + d]; ghn += hk * Whh[k * 192 + 128 + d];
  }
  float r = sigmoidf_(gir + ghr);
  float z = sigmoidf_(giz + ghz);
  float nn = tanhf(gin + r * ghn);
  float hv = (1.0f - z) * nn + z * hrow[nl][d];
  h[n * 64 + d] = hv; out[n * 64 + d] = hv; outH[n * 64 + d] = (half_t)hv;
}

// ---------------------------------------------------------------------------
// BP factor kernel (fused): Z = tanh(gather(out)@Wr + temb + aemb)  [WMMA]
//   P = prod_k Z ;  msg = P @ Wo  [WMMA] ;  atomic scatter to x1[fact[:,0]]
// ---------------------------------------------------------------------------
template <int KMEM, int FPB>
__global__ __launch_bounds__(128) void bp_kernel(const half_t* __restrict__ outH,
                                                 const int* __restrict__ fact,
                                                 const int* __restrict__ ftype,
                                                 const int* __restrict__ atoms,
                                                 const half_t* __restrict__ WrP,
                                                 const float* __restrict__ temb,
                                                 const float* __restrict__ aemb,
                                                 const half_t* __restrict__ WoP,
                                                 float* __restrict__ x1) {
  static_assert(KMEM * FPB <= 32, "rows must fit 2 M-tiles");
  constexpr int ROWS = KMEM * FPB;
  __shared__ half_t Al[32][64];
  __shared__ half_t Zl[32][512];
  __shared__ half_t Pl[16][512];
  __shared__ int ftypeRow[32];
  __shared__ int atomRow[32];
  __shared__ int center[FPB];

  const int tid = threadIdx.x;
  const int f0 = blockIdx.x * FPB;

  {  // stage A rows (gathered f16 node features) + per-row metadata
    int idx = tid * 16;          // 128 threads x 16 halfs = 32 rows x 64
    int row = idx >> 6, col = idx & 63;
    if (row < ROWS) {
      int f = row / KMEM, k = row % KMEM;
      int node = fact[(f0 + f) * KMEM + k];
      if (col == 0) {
        ftypeRow[row] = ftype[(f0 + f) * KMEM + k];
        atomRow[row] = atoms[node];
        if (k == 0) center[f] = node;
      }
      const half_t* srcp = outH + (size_t)node * 64 + col;
      for (int j = 0; j < 16; ++j) Al[row][col + j] = srcp[j];
    } else {
      if (col == 0) { ftypeRow[row] = 0; atomRow[row] = 0; }
      for (int j = 0; j < 16; ++j) Al[row][col + j] = (half_t)0.0f;
    }
  }
  __syncthreads();

  const int w = tid >> 5, lane = tid & 31;
  const int nlo = lane & 15;
  const int mofs = (lane < 16) ? 0 : 8;

  // Z = tanh(A @ Wr + temb + aemb):  2 M-tiles x 32 N-blocks, wave w owns 8 N-blocks
  for (int mt = 0; mt < 2; ++mt) {
    for (int nb8 = 0; nb8 < 8; ++nb8) {
      int nb = w * 8 + nb8;
      v8f acc = {};
#pragma unroll
      for (int kc = 0; kc < 2; ++kc) {
        v16h a = a_frag(&Al[mt * 16][0], 64, kc * 32, lane);
        v16h b = *(const v16h*)(WrP + (size_t)((kc * 32 + nb) * 32 + lane) * 16);
        acc = wmma_f16(a, b, acc);
      }
#pragma unroll
      for (int r = 0; r < 8; ++r) {
        int row = mt * 16 + r + mofs;
        int col = nb * 16 + nlo;
        float v = acc[r] + temb[ftypeRow[row] * 512 + col] + aemb[atomRow[row] * 512 + col];
        Zl[row][col] = (half_t)tanhf(v);
      }
    }
  }
  __syncthreads();

  // P[f, col] = prod_k Z[f*KMEM+k, col]  (rows >= FPB zero-padded)
  for (int idx = tid; idx < 16 * 512; idx += 128) {
    int f = idx >> 9, col = idx & 511;
    float p = 0.0f;
    if (f < FPB) {
      p = 1.0f;
#pragma unroll
      for (int k = 0; k < KMEM; ++k) p *= (float)Zl[f * KMEM + k][col];
    }
    Pl[f][col] = (half_t)p;
  }
  __syncthreads();

  // msg = P @ Wo : 1 M-tile x 4 N-blocks (wave w owns N-block w), K = 512
  // Light double-buffer on the B fragment.
  {
    v8f acc = {};
    v16h bcur = *(const v16h*)(WoP + (size_t)((0 * 4 + w) * 32 + lane) * 16);
#pragma unroll
    for (int kc = 0; kc < 16; ++kc) {
      v16h bnext;
      int kn = (kc + 1 < 16) ? (kc + 1) : 15;
      bnext = *(const v16h*)(WoP + (size_t)((kn * 4 + w) * 32 + lane) * 16);
      v16h a = a_frag(&Pl[0][0], 512, kc * 32, lane);
      acc = wmma_f16(a, bcur, acc);
      bcur = bnext;
    }
#pragma unroll
    for (int r = 0; r < 8; ++r) {
      int f = r + mofs;
      if (f < FPB) {
        int col = w * 16 + nlo;
        atomicAdd(&x1[(size_t)center[f] * 64 + col], acc[r]);
      }
    }
  }
}

// out += relu((mix_w * x1) @ lin_W + lin_b); refresh outH
__global__ __launch_bounds__(256) void mix_kernel(const float* __restrict__ x1,
                                                  const float* __restrict__ mixw,
                                                  const float* __restrict__ linW,
                                                  const float* __restrict__ linb,
                                                  float* __restrict__ out, half_t* __restrict__ outH) {
  __shared__ float yrow[4][64];
  int tid = threadIdx.x, nl = tid >> 6, d = tid & 63;
  int n = blockIdx.x * 4 + nl;
  yrow[nl][d] = mixw[d] * x1[n * 64 + d];
  __syncthreads();
  float acc = linb[d];
  for (int k = 0; k < 64; ++k) acc += yrow[nl][k] * linW[k * 64 + d];
  float v = out[n * 64 + d] + fmaxf(acc, 0.0f);
  out[n * 64 + d] = v; outH[n * 64 + d] = (half_t)v;
}

__global__ void copy_col_kernel(const float* __restrict__ out, float* __restrict__ feat, int off) {
  int i = blockIdx.x * blockDim.x + threadIdx.x;
  if (i >= N_NODES * 64) return;
  int n = i >> 6, d = i & 63;
  feat[(size_t)n * 128 + off + d] = out[i];
}

// ---------------------------------------------------------------------------
// set2set
// ---------------------------------------------------------------------------
__global__ __launch_bounds__(128) void s2s_lstm_kernel(const float* __restrict__ qstar,
                                                       const float* __restrict__ Wih,
                                                       const float* __restrict__ Whh,
                                                       const float* __restrict__ b,
                                                       float* __restrict__ h, float* __restrict__ c) {
  __shared__ float q[256], hr[128];
  int g = blockIdx.x, j = threadIdx.x;
  q[j] = qstar[g * 256 + j];
  q[128 + j] = qstar[g * 256 + 128 + j];
  hr[j] = h[g * 128 + j];
  __syncthreads();
  float gi = b[j], gf = b[128 + j], gg = b[256 + j], go = b[384 + j];
  for (int k = 0; k < 256; ++k) {
    float qk = q[k];
    gi += qk * Wih[k * 512 + j];        gf += qk * Wih[k * 512 + 128 + j];
    gg += qk * Wih[k * 512 + 256 + j];  go += qk * Wih[k * 512 + 384 + j];
  }
  for (int k = 0; k < 128; ++k) {
    float hk = hr[k];
    gi += hk * Whh[k * 512 + j];        gf += hk * Whh[k * 512 + 128 + j];
    gg += hk * Whh[k * 512 + 256 + j];  go += hk * Whh[k * 512 + 384 + j];
  }
  float cv = sigmoidf_(gf) * c[g * 128 + j] + sigmoidf_(gi) * tanhf(gg);
  float hv = sigmoidf_(go) * tanhf(cv);
  c[g * 128 + j] = cv; h[g * 128 + j] = hv;
}

__global__ void s2s_e_kernel(const float* __restrict__ feat, const float* __restrict__ h,
                             const int* __restrict__ batch, float* __restrict__ e,
                             unsigned* __restrict__ emaxk) {
  int n = blockIdx.x * blockDim.x + threadIdx.x;
  if (n >= N_NODES) return;
  int g = batch[n];
  float acc = 0.0f;
  for (int k = 0; k < 128; ++k) acc += feat[(size_t)n * 128 + k] * h[g * 128 + k];
  e[n] = acc;
  atomicMax(&emaxk[g], fkey(acc));
}

__global__ void s2s_a_kernel(const float* __restrict__ e, const unsigned* __restrict__ emaxk,
                             const int* __restrict__ batch, float* __restrict__ a,
                             float* __restrict__ asum) {
  int n = blockIdx.x * blockDim.x + threadIdx.x;
  if (n >= N_NODES) return;
  int g = batch[n];
  float av = expf(e[n] - fkey_inv(emaxk[g]));
  a[n] = av;
  atomicAdd(&asum[g], av);
}

__global__ void s2s_r_kernel(const float* __restrict__ feat, const float* __restrict__ a,
                             const float* __restrict__ asum, const int* __restrict__ batch,
                             float* __restrict__ r) {
  int i = blockIdx.x * blockDim.x + threadIdx.x;
  if (i >= N_NODES * 128) return;
  int n = i >> 7, j = i & 127;
  int g = batch[n];
  atomicAdd(&r[g * 128 + j], (a[n] / asum[g]) * feat[(size_t)n * 128 + j]);
}

__global__ void s2s_q_kernel(const float* __restrict__ h, const float* __restrict__ r,
                             float* __restrict__ qstar) {
  int i = blockIdx.x * blockDim.x + threadIdx.x;
  if (i >= B_GRAPHS * 128) return;
  int g = i >> 7, j = i & 127;
  qstar[g * 256 + j] = h[i];
  qstar[g * 256 + 128 + j] = r[i];
}

__global__ __launch_bounds__(256) void final_kernel(const float* __restrict__ qstar,
                                                    const float* __restrict__ W1,
                                                    const float* __restrict__ b1,
                                                    const float* __restrict__ W2,
                                                    const float* __restrict__ b2,
                                                    float* __restrict__ outp) {
  __shared__ float qs[256], qr[256];
  int g = blockIdx.x, j = threadIdx.x;
  qs[j] = qstar[g * 256 + j];
  __syncthreads();
  float acc = b1[j];
  for (int k = 0; k < 256; ++k) acc += qs[k] * W1[k * 256 + j];
  qr[j] = fmaxf(acc, 0.0f);
  __syncthreads();
  if (j < 12) {
    float o = b2[j];
    for (int k = 0; k < 256; ++k) o += qr[k] * W2[k * 12 + j];
    outp[g * 12 + j] = o;
  }
}

// ---------------------------------------------------------------------------
// host
// ---------------------------------------------------------------------------
extern "C" void kernel_launch(void* const* d_in, const int* in_sizes, int n_in,
                              void* d_out, int out_size, void* d_ws, size_t ws_size,
                              hipStream_t stream) {
  (void)in_sizes; (void)n_in; (void)out_size; (void)ws_size;
  const float* x         = (const float*)d_in[0];
  const float* edge_attr = (const float*)d_in[1];
  const int*   edge_index= (const int*)d_in[2];
  const int*   batch     = (const int*)d_in[3];
  const int*   fact2     = (const int*)d_in[4];
  const int*   ftype2    = (const int*)d_in[5];
  const int*   fact3     = (const int*)d_in[6];
  const int*   ftype3    = (const int*)d_in[7];
  const int*   fact4     = (const int*)d_in[8];
  const int*   ftype4    = (const int*)d_in[9];
  const float* lin0_W = (const float*)d_in[10];
  const float* lin0_b = (const float*)d_in[11];
  const float* nn1_W  = (const float*)d_in[12];
  const float* nn1_b  = (const float*)d_in[13];
  const float* nn2_W  = (const float*)d_in[14];
  const float* nn2_b  = (const float*)d_in[15];
  const float* root_W = (const float*)d_in[16];
  const float* root_b = (const float*)d_in[17];
  const float* gru_Wih= (const float*)d_in[18];
  const float* gru_Whh= (const float*)d_in[19];
  const float* gru_bih= (const float*)d_in[20];
  const float* gru_bhh= (const float*)d_in[21];
  const float* bp_Wr  = (const float*)d_in[22];
  const float* bp_temb= (const float*)d_in[23];
  const float* bp_aemb= (const float*)d_in[24];
  const float* bp_Wo  = (const float*)d_in[25];
  const float* lin_W  = (const float*)d_in[26];
  const float* lin_b  = (const float*)d_in[27];
  const float* mix_w  = (const float*)d_in[28];
  const float* s2s_Wih= (const float*)d_in[29];
  const float* s2s_Whh= (const float*)d_in[30];
  const float* s2s_b  = (const float*)d_in[31];
  const float* lin1_W = (const float*)d_in[32];
  const float* lin1_b = (const float*)d_in[33];
  const float* lin2_W = (const float*)d_in[34];
  const float* lin2_b = (const float*)d_in[35];

  char* wsb = (char*)d_ws;
  size_t off = 0;
  auto alloc = [&](size_t bytes) -> void* {
    void* p = wsb + off;
    off = (off + bytes + 255) & ~(size_t)255;
    return p;
  };
  float*  out   = (float*)alloc((size_t)N_NODES * 64 * 4);
  float*  h     = (float*)alloc((size_t)N_NODES * 64 * 4);
  half_t* outH  = (half_t*)alloc((size_t)N_NODES * 64 * 2);
  half_t* h1H   = (half_t*)alloc((size_t)E_EDGES * 128 * 2);
  half_t* Tp    = (half_t*)alloc((size_t)258 * 4 * 32 * 16 * 2);
  half_t* WrP   = (half_t*)alloc((size_t)3 * 32768 * 2);
  half_t* WoP   = (half_t*)alloc((size_t)3 * 32768 * 2);
  float*  s     = (float*)alloc((size_t)N_NODES * 64 * 4);
  float*  cnt   = (float*)alloc((size_t)N_NODES * 4);
  float*  x1    = (float*)alloc((size_t)N_NODES * 64 * 4);
  float*  feat  = (float*)alloc((size_t)N_NODES * 128 * 4);
  int*    atoms = (int*)alloc((size_t)N_NODES * 4);
  float*  ev    = (float*)alloc((size_t)N_NODES * 4);
  float*  av    = (float*)alloc((size_t)N_NODES * 4);
  float*  hB    = (float*)alloc((size_t)B_GRAPHS * 128 * 4);
  float*  cB    = (float*)alloc((size_t)B_GRAPHS * 128 * 4);
  float*  qstar = (float*)alloc((size_t)B_GRAPHS * 256 * 4);
  unsigned* emaxk = (unsigned*)alloc((size_t)B_GRAPHS * 4);
  float*  asum  = (float*)alloc((size_t)B_GRAPHS * 4);
  float*  rB    = (float*)alloc((size_t)B_GRAPHS * 128 * 4);

  // ---- setup (recomputed every call; deterministic) ----
  fill_kernel<<<(N_NODES + 255) / 256, 256, 0, stream>>>(cnt, 0.0f, N_NODES);
  count_kernel<<<(E_EDGES + 255) / 256, 256, 0, stream>>>(edge_index, cnt);
  prep_kernel<<<N_NODES / 4, 256, 0, stream>>>(x, lin0_W, lin0_b, out, h, outH, atoms);
  edge_h1_kernel<<<E_EDGES / 2, 256, 0, stream>>>(edge_attr, nn1_W, nn1_b, h1H);
  {
    int tot = 258 * 4 * 32 * 16;
    pack_T_kernel<<<(tot + 255) / 256, 256, 0, stream>>>(nn2_W, nn2_b, Tp);
  }
  for (int kk = 0; kk < 3; ++kk) {
    pack_B_kernel<<<(32768 + 255) / 256, 256, 0, stream>>>(bp_Wr + (size_t)kk * 64 * 512, 64, 512,
                                                           WrP + (size_t)kk * 32768);
    pack_B_kernel<<<(32768 + 255) / 256, 256, 0, stream>>>(bp_Wo + (size_t)kk * 512 * 64, 512, 64,
                                                           WoP + (size_t)kk * 32768);
  }

  // ---- 3 x (NNConv + GRU) ----
  for (int it = 0; it < 3; ++it) {
    fill_kernel<<<(N_NODES * 64 + 255) / 256, 256, 0, stream>>>(s, 0.0f, N_NODES * 64);
    conv_msg_kernel<<<E_EDGES / 64, 128, 0, stream>>>(outH, h1H, Tp, edge_index, s);
    gru_kernel<<<N_NODES / 4, 256, 0, stream>>>(root_W, root_b, gru_Wih, gru_Whh, gru_bih, gru_bhh,
                                                s, cnt, out, h, outH);
  }
  copy_col_kernel<<<(N_NODES * 64 + 255) / 256, 256, 0, stream>>>(out, feat, 0);

  // ---- 3 x BP rounds over 3 factor orders ----
  for (int kk = 0; kk < 3; ++kk) {
    fill_kernel<<<(N_NODES * 64 + 255) / 256, 256, 0, stream>>>(x1, 0.0f, N_NODES * 64);
    bp_kernel<3, 8><<<F2_N / 8, 128, 0, stream>>>(outH, fact2, ftype2, atoms,
        WrP + (size_t)kk * 32768, bp_temb + (size_t)kk * NT_TYPES * 512,
        bp_aemb + (size_t)kk * NA_TYPES * 512, WoP + (size_t)kk * 32768, x1);
    bp_kernel<4, 8><<<F3_N / 8, 128, 0, stream>>>(outH, fact3, ftype3, atoms,
        WrP + (size_t)kk * 32768, bp_temb + (size_t)kk * NT_TYPES * 512,
        bp_aemb + (size_t)kk * NA_TYPES * 512, WoP + (size_t)kk * 32768, x1);
    bp_kernel<5, 4><<<F4_N / 4, 128, 0, stream>>>(outH, fact4, ftype4, atoms,
        WrP + (size_t)kk * 32768, bp_temb + (size_t)kk * NT_TYPES * 512,
        bp_aemb + (size_t)kk * NA_TYPES * 512, WoP + (size_t)kk * 32768, x1);
    mix_kernel<<<N_NODES / 4, 256, 0, stream>>>(x1, mix_w, lin_W, lin_b, out, outH);
  }
  copy_col_kernel<<<(N_NODES * 64 + 255) / 256, 256, 0, stream>>>(out, feat, 64);

  // ---- set2set (3 steps) ----
  fill_kernel<<<(B_GRAPHS * 128 + 255) / 256, 256, 0, stream>>>(hB, 0.0f, B_GRAPHS * 128);
  fill_kernel<<<(B_GRAPHS * 128 + 255) / 256, 256, 0, stream>>>(cB, 0.0f, B_GRAPHS * 128);
  fill_kernel<<<(B_GRAPHS * 256 + 255) / 256, 256, 0, stream>>>(qstar, 0.0f, B_GRAPHS * 256);
  for (int t = 0; t < 3; ++t) {
    fill_kernel<<<(B_GRAPHS + 255) / 256, 256, 0, stream>>>((float*)emaxk, 0.0f, B_GRAPHS);
    fill_kernel<<<(B_GRAPHS + 255) / 256, 256, 0, stream>>>(asum, 0.0f, B_GRAPHS);
    fill_kernel<<<(B_GRAPHS * 128 + 255) / 256, 256, 0, stream>>>(rB, 0.0f, B_GRAPHS * 128);
    s2s_lstm_kernel<<<B_GRAPHS, 128, 0, stream>>>(qstar, s2s_Wih, s2s_Whh, s2s_b, hB, cB);
    s2s_e_kernel<<<(N_NODES + 255) / 256, 256, 0, stream>>>(feat, hB, batch, ev, emaxk);
    s2s_a_kernel<<<(N_NODES + 255) / 256, 256, 0, stream>>>(ev, emaxk, batch, av, asum);
    s2s_r_kernel<<<(N_NODES * 128 + 255) / 256, 256, 0, stream>>>(feat, av, asum, batch, rB);
    s2s_q_kernel<<<(B_GRAPHS * 128 + 255) / 256, 256, 0, stream>>>(hB, rB, qstar);
  }
  final_kernel<<<B_GRAPHS, 256, 0, stream>>>(qstar, lin1_W, lin1_b, lin2_W, lin2_b, (float*)d_out);
}
